// EpiSIGNetV3_NoSIG_15204184227951
// MI455X (gfx1250) — compile-verified, matrix operand order checked by reading, not run on hardware
//
#include <hip/hip_runtime.h>
#include <hip/hip_bf16.h>
#include <math.h>

// ---------------------------------------------------------------------------
// EpiSIGNetV3 forward for MI455X (gfx1250, wave32).
// All large GEMMs: v_wmma_f32_16x16x32_f16 with f32 accumulate; weight tiles
// staged to LDS with double-buffered global_load_async_to_lds_b128 (ASYNCcnt);
// LN/softmax/GRU gate math in fp32 VALU. Compute-bound (~36 GFLOP, ~6MB data).
// ---------------------------------------------------------------------------

typedef _Float16 half_t;
typedef __attribute__((ext_vector_type(16))) _Float16 v16h;
typedef __attribute__((ext_vector_type(8)))  _Float16 h8v;
typedef __attribute__((ext_vector_type(8)))  float    v8f;

#define B_     16
#define T_     64
#define N_     512
#define HID_   256
#define HEADS_ 4
#define HD_    64
#define FC_    16
#define HOR_   24
#define H2_    128
#define H4_    64
#define ROWS_  (B_ * N_)   // 8192

// LDS byte-offset of a __shared__ object (generic -> AS3 -> u32).
__device__ __forceinline__ unsigned lds_off_u32(const void* p) {
  return (unsigned)(unsigned long long)(__attribute__((address_space(3))) const char*)p;
}

// Async copy 16B global -> LDS (tracked by ASYNCcnt, no VGPR staging).
__device__ __forceinline__ void async_g2l_b128(unsigned lds_byte_off, const void* gptr) {
  asm volatile("global_load_async_to_lds_b128 %0, %1, off"
               :: "v"(lds_byte_off), "v"(gptr) : "memory");
}
__device__ __forceinline__ void wait_asynccnt0() {
  asm volatile("s_wait_asynccnt 0x0" ::: "memory");
}

// ---------------- WMMA fragment helpers (layout per ISA 7.12.2) ------------
// A-frag: 16x32 f16 tile, row-major source, lane L holds row m=L%16,
//         K halves: k = (e/8)*16 + (L/16)*8 + ((e/2)%4)*2 + (e&1)
// B-frag: same loader applied to a [16(N) x K] row-major tile (mirror layout).
__device__ __forceinline__ v16h load_frag16(const half_t* p0, int ld, int lane) {
  const half_t* p = p0 + (size_t)(lane & 15) * ld + ((lane >> 4) << 3);
  h8v lo = *(const h8v*)(p);
  h8v hi = *(const h8v*)(p + 16);
  v16h f;
#pragma unroll
  for (int i = 0; i < 8; ++i) { f[i] = lo[i]; f[i + 8] = hi[i]; }
  return f;
}

__device__ __forceinline__ v8f wmma_accum(const half_t* A, int lda,
                                          const half_t* Bw, int ldb,
                                          int K, int lane, v8f acc) {
  for (int k0 = 0; k0 < K; k0 += 32) {
    v16h af = load_frag16(A  + k0, lda, lane);
    v16h bf = load_frag16(Bw + k0, ldb, lane);
    acc = __builtin_amdgcn_wmma_f32_16x16x32_f16(false, af, false, bf,
                                                 (short)0, acc, false, false);
  }
  return acc;
}

__device__ __forceinline__ float sigm(float v) { return 1.f / (1.f + __expf(-v)); }

// ---------------------------- fp32 -> fp16 ---------------------------------
__global__ void k_f2h(const float* __restrict__ s, half_t* __restrict__ d, int n) {
  int i = blockIdx.x * 256 + threadIdx.x;
  if (i < n) d[i] = (half_t)s[i];
}

// --------------------------- conv front-end --------------------------------
struct ConvP {
  const float *fdw, *fdb, *fg1, *fb1, *fpw, *fpb, *fg2, *fb2;
  const float *sdw[3], *sdb[3], *sg1[3], *sb1[3], *spw[3], *spb[3], *sg2[3], *sb2[3];
  const float *scale_w;
};

__global__ void k_conv(const float* __restrict__ x, ConvP p,
                       half_t* __restrict__ fusedH) {
  __shared__ float xb[T_];
  __shared__ float feat[FC_][T_];
  __shared__ float yb[FC_][T_];
  int row = blockIdx.x;                 // b*N + node
  int b = row >> 9, node = row & 511;
  int t = threadIdx.x;                  // 64 threads
  xb[t] = x[((size_t)b * T_ + t) * N_ + node];
  __syncthreads();

  // fc: 1 -> 16 channels
  float xm = (t > 0)  ? xb[t - 1] : 0.f;
  float xc = xb[t];
  float xp = (t < 63) ? xb[t + 1] : 0.f;
  float y0 = xm * p.fdw[0] + xc * p.fdw[1] + xp * p.fdw[2] + p.fdb[0];
  y0 = y0 * p.fg1[0] + p.fb1[0];
  y0 = y0 > 0.f ? y0 : 0.f;
#pragma unroll
  for (int c = 0; c < FC_; ++c) {
    float v = (y0 * p.fpw[c] + p.fpb[c]) * p.fg2[c] + p.fb2[c];
    feat[c][t] = v > 0.f ? v : 0.f;
  }
  __syncthreads();

  float s0 = p.scale_w[0], s1 = p.scale_w[1], s2 = p.scale_w[2];
  float smx = fmaxf(s0, fmaxf(s1, s2));
  float e0 = __expf(s0 - smx), e1 = __expf(s1 - smx), e2 = __expf(s2 - smx);
  float sinv = 1.f / (e0 + e1 + e2);
  float sw[3] = { e0 * sinv, e1 * sinv, e2 * sinv };

  float acc[FC_];
#pragma unroll
  for (int o = 0; o < FC_; ++o) acc[o] = 0.f;

  for (int br = 0; br < 3; ++br) {
    int dil = 1 << br;
#pragma unroll
    for (int c = 0; c < FC_; ++c) {
      float a0 = (t - dil >= 0)  ? feat[c][t - dil] : 0.f;
      float a1 = feat[c][t];
      float a2 = (t + dil < T_)  ? feat[c][t + dil] : 0.f;
      float v = a0 * p.sdw[br][c * 3 + 0] + a1 * p.sdw[br][c * 3 + 1]
              + a2 * p.sdw[br][c * 3 + 2] + p.sdb[br][c];
      v = v * p.sg1[br][c] + p.sb1[br][c];
      yb[c][t] = v > 0.f ? v : 0.f;
    }
    __syncthreads();
#pragma unroll
    for (int o = 0; o < FC_; ++o) {
      float s = 0.f;
#pragma unroll
      for (int c = 0; c < FC_; ++c) s += yb[c][t] * p.spw[br][o * FC_ + c];
      float v = (s + p.spb[br][o]) * p.sg2[br][o] + p.sb2[br][o];
      acc[o] += sw[br] * (v > 0.f ? v : 0.f);
    }
    __syncthreads();
  }
#pragma unroll
  for (int o = 0; o < FC_; ++o)
    fusedH[(size_t)row * (FC_ * T_) + o * T_ + t] = (half_t)acc[o];
}

// ------------------ macro-tile WMMA GEMM with async LDS B ------------------
// C[M x N] = A[M x K](f16) @ W[N x K]^T(f16) + bias.
// block = 256 (8 waves) -> 128(M) x 64(N) macro-tile; wave w owns rows
// [w*16, w*16+16) and 4 accumulators across the 64 columns, reusing its
// A fragment 4x. The 64x32 B tile is double-buffered in LDS and filled with
// global_load_async_to_lds_b128 (each thread copies 16B), overlapping the
// copy of step k+1 with the 32 WMMAs of step k.
// mode: 0 = f32 out, 1 = f32 out + relu, 2 = qkv scatter.
__global__ void k_gemm(const half_t* __restrict__ A, const half_t* __restrict__ W,
                       const float* __restrict__ bias, float* __restrict__ C,
                       const float* __restrict__ geo_w, half_t* __restrict__ qH,
                       half_t* __restrict__ kH, half_t* __restrict__ vTH,
                       int N, int K, int mode) {
  __shared__ alignas(16) half_t bbuf[2][64][32];   // 2 x 4KB
  int tid = threadIdx.x, lane = tid & 31, wave = tid >> 5;
  int n0 = blockIdx.x * 64;
  int m0 = blockIdx.y * 128 + wave * 16;
  int sn = tid >> 2, sk = (tid & 3) << 3;          // stage: row, 8-half chunk

  const half_t* Ab = A + (size_t)m0 * K;
  const half_t* Wg = W + (size_t)(n0 + sn) * K + sk;

  // prologue: stage k0 = 0 into buffer 0
  async_g2l_b128(lds_off_u32(&bbuf[0][sn][sk]), Wg);

  v8f acc[4];
#pragma unroll
  for (int nt = 0; nt < 4; ++nt) acc[nt] = (v8f){};

  int cur = 0;
  for (int k0 = 0; k0 < K; k0 += 32) {
    wait_asynccnt0();          // our async fills of bbuf[cur] are done
    __syncthreads();           // ... and visible to / done by all waves
    if (k0 + 32 < K) {
      async_g2l_b128(lds_off_u32(&bbuf[cur ^ 1][sn][sk]), Wg + k0 + 32);
      __builtin_prefetch(Ab + k0 + 32, 0, 3);
    }
    v16h af = load_frag16(Ab + k0, K, lane);
#pragma unroll
    for (int nt = 0; nt < 4; ++nt) {
      v16h bf = load_frag16(&bbuf[cur][nt * 16][0], 32, lane);
      acc[nt] = __builtin_amdgcn_wmma_f32_16x16x32_f16(false, af, false, bf,
                                                       (short)0, acc[nt], false, false);
    }
    __syncthreads();           // all reads of bbuf[cur] done before next overwrite
    cur ^= 1;
  }

  int mb = m0 + ((lane >> 4) << 3);
  if (mode <= 1) {
#pragma unroll
    for (int nt = 0; nt < 4; ++nt) {
      int n = n0 + nt * 16 + (lane & 15);
      float bs = bias ? bias[n] : 0.f;
#pragma unroll
      for (int r = 0; r < 8; ++r) {
        float v = acc[nt][r] + bs;
        if (mode == 1) v = v > 0.f ? v : 0.f;
        C[(size_t)(mb + r) * N + n] = v;
      }
    }
  } else {
    float geo = sigm(geo_w[0]);
    float qscale = (1.f - geo) * 0.125f;           // (1-geo)/sqrt(HD)
#pragma unroll
    for (int nt = 0; nt < 4; ++nt) {
      int j = n0 + nt * 16 + (lane & 15);
      int s = j >> 8, h = (j >> 6) & 3, d = j & 63;
      float bs = bias[j];
#pragma unroll
      for (int r = 0; r < 8; ++r) {
        int token = mb + r;
        int b = token >> 9, node = token & 511;
        size_t bh = (size_t)(b * HEADS_ + h);
        float v = acc[nt][r] + bs;
        if (s == 0)      qH[(bh * N_ + node) * HD_ + d] = (half_t)(v * qscale);
        else if (s == 1) kH[(bh * N_ + node) * HD_ + d] = (half_t)v;
        else             vTH[(bh * HD_ + d) * N_ + node] = (half_t)v;
      }
    }
  }
}

// ------------------------------ LayerNorm ----------------------------------
__global__ void k_ln(const float* __restrict__ X, const float* __restrict__ resid,
                     const float* __restrict__ g, const float* __restrict__ bb,
                     float* __restrict__ outF, half_t* __restrict__ outH, int relu) {
  __shared__ float red[HID_];
  __shared__ float stat[2];
  int row = blockIdx.x, t = threadIdx.x;   // 256 threads
  float v = X[(size_t)row * HID_ + t];
  if (resid) v += resid[(size_t)row * HID_ + t];
  red[t] = v; __syncthreads();
  for (int s = 128; s > 0; s >>= 1) { if (t < s) red[t] += red[t + s]; __syncthreads(); }
  if (t == 0) stat[0] = red[0] * (1.f / HID_);
  __syncthreads();
  float m = stat[0];
  float d = v - m;
  red[t] = d * d; __syncthreads();
  for (int s = 128; s > 0; s >>= 1) { if (t < s) red[t] += red[t + s]; __syncthreads(); }
  if (t == 0) stat[1] = red[0] * (1.f / HID_);
  __syncthreads();
  float y = d * rsqrtf(stat[1] + 1e-5f) * g[t] + bb[t];
  if (relu) y = y > 0.f ? y : 0.f;
  outF[(size_t)row * HID_ + t] = y;
  outH[(size_t)row * HID_ + t] = (half_t)y;
}

// --------------------- fused attention per (b,h,32 rows) -------------------
#define SMEM_ATTN (32 * 512 * 4 + 32 * 512 * 2 + 256 * 4 + 64 * 4)
__global__ void k_attn(const half_t* __restrict__ qH, const half_t* __restrict__ kH,
                       const half_t* __restrict__ vTH, const float* __restrict__ adj,
                       const float* __restrict__ geo_w, half_t* __restrict__ attOutH) {
  extern __shared__ char smem[];
  float*  scb = (float*)smem;                                 // 32*512 f32
  half_t* pb  = (half_t*)(smem + 32 * 512 * 4);               // 32*512 f16
  float*  red = (float*)(smem + 32 * 512 * 4 + 32 * 512 * 2); // 256 f32
  float*  rowmax = red + 256;
  float*  rowsum = rowmax + 32;

  int bh = blockIdx.x;                // b*4 + h
  int rb = blockIdx.y;                // 32-row query block
  int b  = bh >> 2, h = bh & 3;
  int lane = threadIdx.x & 31, wave = threadIdx.x >> 5;
  float g5 = sigm(geo_w[0]) * 5.f;

  const half_t* Qb = qH  + ((size_t)bh * N_ + rb * 32) * HD_;
  const half_t* Kb = kH  + (size_t)bh * N_ * HD_;
  const half_t* Vt = vTH + (size_t)bh * HD_ * N_;

  // Phase 1: 32x512 scores = 2x32 tiles of 16x16, 8 per wave (K=64).
#pragma unroll
  for (int i = 0; i < 8; ++i) {
    int tt = wave * 8 + i;
    int mt = tt & 1, nt = tt >> 1;
    v8f acc = {};
    acc = wmma_accum(Qb + (size_t)mt * 16 * HD_, HD_,
                     Kb + (size_t)nt * 16 * HD_, HD_, HD_, lane, acc);
    int key = nt * 16 + (lane & 15);
    int mb  = mt * 16 + ((lane >> 4) << 3);
#pragma unroll
    for (int r = 0; r < 8; ++r) {
      int m = mb + r;
      int qnode = rb * 32 + m;
      scb[m * N_ + key] = acc[r] + g5 * adj[(size_t)qnode * N_ + key];
    }
  }
  __syncthreads();

  // Phase 2: row-wise softmax (8 threads per row).
  {
    int row = threadIdx.x >> 3, sub = threadIdx.x & 7;
    float* srow = scb + row * N_;
    float mx = -3.0e38f;
    for (int c = sub; c < N_; c += 8) mx = fmaxf(mx, srow[c]);
    red[row * 8 + sub] = mx; __syncthreads();
    if (sub == 0) {
      float m2 = red[row * 8];
#pragma unroll
      for (int j2 = 1; j2 < 8; ++j2) m2 = fmaxf(m2, red[row * 8 + j2]);
      rowmax[row] = m2;
    }
    __syncthreads();
    mx = rowmax[row];
    half_t* prow = pb + row * N_;
    float sm = 0.f;
    for (int c = sub; c < N_; c += 8) {
      float e = __expf(srow[c] - mx);
      prow[c] = (half_t)e;
      sm += e;
    }
    red[row * 8 + sub] = sm; __syncthreads();
    if (sub == 0) {
      float s2 = 0.f;
#pragma unroll
      for (int j2 = 0; j2 < 8; ++j2) s2 += red[row * 8 + j2];
      rowsum[row] = s2;
    }
    __syncthreads();
    float inv = 1.f / rowsum[row];
    for (int c = sub; c < N_; c += 8) prow[c] = (half_t)((float)prow[c] * inv);
  }
  __syncthreads();

  // Phase 3: out 32x64 = 2x4 tiles, one per wave. A = probs (LDS), B = V^T.
  {
    int mt = wave & 1, nt = wave >> 1;
    v8f acc = {};
    acc = wmma_accum(pb + (size_t)mt * 16 * N_, N_,
                     Vt + (size_t)nt * 16 * N_, N_, N_, lane, acc);
    int dcol = nt * 16 + (lane & 15);
    int mb   = mt * 16 + ((lane >> 4) << 3);
#pragma unroll
    for (int r = 0; r < 8; ++r) {
      int node = rb * 32 + mb + r;
      int token = b * N_ + node;
      attOutH[(size_t)token * HID_ + h * HD_ + dcol] = (half_t)acc[r];
    }
  }
}

// ------------------------------- GRU decoder -------------------------------
// One workgroup (128 threads, 4 waves) per 16-row tile; 24 sequential steps;
// whh and op1 GEMMs via WMMA from LDS-resident f16 hidden state.
__global__ void k_gru(const float* __restrict__ h0F, const float* __restrict__ x,
                      const half_t* __restrict__ whhH, const half_t* __restrict__ op1WH,
                      const float* __restrict__ wih, const float* __restrict__ bih,
                      const float* __restrict__ bhh, const float* __restrict__ op1b,
                      const float* __restrict__ op2w, const float* __restrict__ op2b,
                      float* __restrict__ gruPred) {
  __shared__ alignas(16) float  hF[16][H2_];
  __shared__ alignas(16) half_t hH[16][H2_];
  __shared__ alignas(16) float  gh[16][3 * H2_];
  __shared__ alignas(16) half_t o1H[16][H4_];
  __shared__ float xin[16];
  __shared__ float pr[16][8];

  int tid = threadIdx.x, lane = tid & 31, wave = tid >> 5;
  int r0 = blockIdx.x * 16;

  for (int idx = tid; idx < 16 * H2_; idx += 128) {
    int m = idx >> 7, j = idx & 127;
    float v = h0F[(size_t)(r0 + m) * H2_ + j];
    hF[m][j] = v; hH[m][j] = (half_t)v;
  }
  if (tid < 16) {
    int token = r0 + tid, b = token >> 9, node = token & 511;
    xin[tid] = x[((size_t)b * T_ + (T_ - 1)) * N_ + node];   // x_last
  }
  __syncthreads();

  for (int step = 0; step < HOR_; ++step) {
    // gh = h @ whh^T : N=384 -> 24 tiles, 6 per wave, K=128.
    for (int t6 = 0; t6 < 6; ++t6) {
      int nt = wave * 6 + t6;
      v8f acc = {};
      acc = wmma_accum(&hH[0][0], H2_, whhH + (size_t)nt * 16 * H2_, H2_,
                       H2_, lane, acc);
      int col = nt * 16 + (lane & 15);
      int mb  = (lane >> 4) << 3;
      float bs = bhh[col];
#pragma unroll
      for (int r = 0; r < 8; ++r) gh[mb + r][col] = acc[r] + bs;
    }
    __syncthreads();

    // gates + hidden update (fp32).
#pragma unroll
    for (int it = 0; it < 16; ++it) {
      int idx = tid + it * 128;
      int m = idx >> 7, j = idx & 127;
      float xi = xin[m];
      float rg = sigm(xi * wih[j]        + bih[j]        + gh[m][j]);
      float zg = sigm(xi * wih[H2_ + j]  + bih[H2_ + j]  + gh[m][H2_ + j]);
      float ng = tanhf(xi * wih[2*H2_+j] + bih[2*H2_+j]  + rg * gh[m][2*H2_ + j]);
      float hn = (1.f - zg) * ng + zg * hF[m][j];
      hF[m][j] = hn; hH[m][j] = (half_t)hn;
    }
    __syncthreads();

    // op1 = relu(h @ op1_w^T + b): 4 tiles, one per wave (K=128).
    {
      v8f acc = {};
      acc = wmma_accum(&hH[0][0], H2_, op1WH + (size_t)wave * 16 * H2_, H2_,
                       H2_, lane, acc);
      int col = wave * 16 + (lane & 15);
      int mb  = (lane >> 4) << 3;
      float bs = op1b[col];
#pragma unroll
      for (int r = 0; r < 8; ++r) {
        float v = acc[r] + bs;
        o1H[mb + r][col] = (half_t)(v > 0.f ? v : 0.f);
      }
    }
    __syncthreads();

    // op2: per-row dot(64); 8 threads per row.
    {
      int m = tid >> 3, sub = tid & 7;
      float s = 0.f;
#pragma unroll
      for (int j = sub * 8; j < sub * 8 + 8; ++j)
        s += (float)o1H[m][j] * op2w[j];
      pr[m][sub] = s; __syncthreads();
      if (sub == 0) {
        float p = op2b[0];
#pragma unroll
        for (int j = 0; j < 8; ++j) p += pr[m][j];
        gruPred[(size_t)(r0 + m) * HOR_ + step] = p;
        xin[m] = p;
      }
    }
    __syncthreads();
  }
}

// ------------------------- final gate/decay/hw combine ---------------------
__global__ void k_final(const float* __restrict__ x, const float* __restrict__ g1F,
                        const float* __restrict__ g2w, const float* __restrict__ g2b,
                        const float* __restrict__ gruPred, const float* __restrict__ log_decay,
                        const float* __restrict__ hww, const float* __restrict__ hwb,
                        const float* __restrict__ hw_ratio, float* __restrict__ out) {
  int token = blockIdx.x * blockDim.x + threadIdx.x;
  if (token >= ROWS_) return;
  int b = token >> 9, node = token & 511;
  float ratio = sigm(hw_ratio[0]);
  float lam = __expf(log_decay[0]);
  float xl = x[((size_t)b * T_ + (T_ - 1)) * N_ + node];
  float hx[4];
#pragma unroll
  for (int j = 0; j < 4; ++j)
    hx[j] = x[((size_t)b * T_ + (T_ - 4 + j)) * N_ + node];
  const float* gr = g1F + (size_t)token * H4_;
  for (int t = 0; t < HOR_; ++t) {
    float gs = g2b[t];
    for (int j = 0; j < H4_; ++j) gs += gr[j] * g2w[t * H4_ + j];
    float gate = sigm(gs);
    float dp = xl * __expf(-lam * (float)(t + 1));
    float f = gate * gruPred[(size_t)token * HOR_ + t] + (1.f - gate) * dp;
    float ho = hwb[t];
#pragma unroll
    for (int j = 0; j < 4; ++j) ho += hx[j] * hww[t * 4 + j];
    f = ratio * f + (1.f - ratio) * ho;
    out[((size_t)b * HOR_ + t) * N_ + node] = f;   // (B, HORIZON, N)
  }
  if (token == 0) out[(size_t)B_ * HOR_ * N_] = 0.f;  // tuple's scalar 0.0
}

// ---------------------------------------------------------------------------
extern "C" void kernel_launch(void* const* d_in, const int* in_sizes, int n_in,
                              void* d_out, int out_size, void* d_ws, size_t ws_size,
                              hipStream_t stream) {
  (void)in_sizes; (void)n_in; (void)out_size; (void)ws_size;
  auto F = [&](int i) { return (const float*)d_in[i]; };

  // Inputs flattened depth-first in setup_inputs() dict order.
  const float* x   = F(0);
  const float* adj = F(1);
  const float* scale_w = F(34);
  const float* proj_w = F(35); const float* proj_b = F(36);
  const float* ln_g = F(37);   const float* ln_b = F(38);
  const float* qkv_w = F(39);  const float* qkv_b = F(40);
  const float* ao_w = F(41);   const float* ao_b = F(42);
  const float* an_g = F(43);   const float* an_b = F(44);
  const float* geo_w = F(45);
  const float* gru_wih = F(46); const float* gru_whh = F(47);
  const float* gru_bih = F(48); const float* gru_bhh = F(49);
  const float* hp_w = F(50);   const float* hp_b = F(51);
  const float* op1_w = F(52);  const float* op1_b = F(53);
  const float* op2_w = F(54);  const float* op2_b = F(55);
  const float* log_decay = F(56);
  const float* g1_w = F(57);   const float* g1_b = F(58);
  const float* g2_w = F(59);   const float* g2_b = F(60);
  const float* hw_w = F(61);   const float* hw_b = F(62);
  const float* hw_ratio = F(63);

  // Workspace carve-up (256B aligned slices).
  char* wsp = (char*)d_ws;
  auto alloc = [&](size_t bytes) {
    void* p = (void*)wsp;
    wsp += (bytes + 255) & ~(size_t)255;
    return p;
  };
  half_t* fusedH  = (half_t*)alloc((size_t)ROWS_ * FC_ * T_ * 2);
  half_t* projWH  = (half_t*)alloc((size_t)HID_ * FC_ * T_ * 2);
  half_t* qkvWH   = (half_t*)alloc((size_t)3 * HID_ * HID_ * 2);
  half_t* aoWH    = (half_t*)alloc((size_t)HID_ * HID_ * 2);
  half_t* hpWH    = (half_t*)alloc((size_t)H2_ * HID_ * 2);
  half_t* g1WH    = (half_t*)alloc((size_t)H4_ * HID_ * 2);
  half_t* whhWH   = (half_t*)alloc((size_t)3 * H2_ * H2_ * 2);
  half_t* op1WH   = (half_t*)alloc((size_t)H4_ * H2_ * 2);
  float*  projTmp = (float*)alloc((size_t)ROWS_ * HID_ * 4);
  float*  featsF  = (float*)alloc((size_t)ROWS_ * HID_ * 4);
  half_t* featsH  = (half_t*)alloc((size_t)ROWS_ * HID_ * 2);
  half_t* qH      = (half_t*)alloc((size_t)B_ * HEADS_ * N_ * HD_ * 2);
  half_t* kH      = (half_t*)alloc((size_t)B_ * HEADS_ * N_ * HD_ * 2);
  half_t* vTH     = (half_t*)alloc((size_t)B_ * HEADS_ * HD_ * N_ * 2);
  half_t* attOutH = (half_t*)alloc((size_t)ROWS_ * HID_ * 2);
  float*  aoTmp   = (float*)alloc((size_t)ROWS_ * HID_ * 4);
  float*  feats2F = (float*)alloc((size_t)ROWS_ * HID_ * 4);
  half_t* feats2H = (half_t*)alloc((size_t)ROWS_ * HID_ * 2);
  float*  h0F     = (float*)alloc((size_t)ROWS_ * H2_ * 4);
  float*  g1F     = (float*)alloc((size_t)ROWS_ * H4_ * 4);
  float*  gruPred = (float*)alloc((size_t)ROWS_ * HOR_ * 4);

  // 1) weight fp32->fp16 packs
  struct { const float* s; half_t* d; int n; } cv[] = {
    { proj_w,  projWH, HID_ * FC_ * T_ },
    { qkv_w,   qkvWH,  3 * HID_ * HID_ },
    { ao_w,    aoWH,   HID_ * HID_ },
    { hp_w,    hpWH,   H2_ * HID_ },
    { g1_w,    g1WH,   H4_ * HID_ },
    { gru_whh, whhWH,  3 * H2_ * H2_ },
    { op1_w,   op1WH,  H4_ * H2_ },
  };
  for (auto& c : cv)
    k_f2h<<<(c.n + 255) / 256, 256, 0, stream>>>(c.s, c.d, c.n);

  // 2) conv front-end
  ConvP cp;
  cp.fdw = F(2); cp.fdb = F(3); cp.fg1 = F(4); cp.fb1 = F(5);
  cp.fpw = F(6); cp.fpb = F(7); cp.fg2 = F(8); cp.fb2 = F(9);
  for (int i = 0; i < 3; ++i) {
    int base = 10 + 8 * i;
    cp.sdw[i] = F(base + 0); cp.sdb[i] = F(base + 1);
    cp.sg1[i] = F(base + 2); cp.sb1[i] = F(base + 3);
    cp.spw[i] = F(base + 4); cp.spb[i] = F(base + 5);
    cp.sg2[i] = F(base + 6); cp.sb2[i] = F(base + 7);
  }
  cp.scale_w = scale_w;
  k_conv<<<ROWS_, T_, 0, stream>>>(x, cp, fusedH);

  // 3) proj GEMM + LN/ReLU
  k_gemm<<<dim3(HID_ / 64, ROWS_ / 128), 256, 0, stream>>>(
      fusedH, projWH, proj_b, projTmp, nullptr, nullptr, nullptr, nullptr,
      HID_, FC_ * T_, 0);
  k_ln<<<ROWS_, HID_, 0, stream>>>(projTmp, nullptr, ln_g, ln_b, featsF, featsH, 1);

  // 4) QKV GEMM (scatter to q/k/v^T, q pre-scaled by (1-geo)/8)
  k_gemm<<<dim3(3 * HID_ / 64, ROWS_ / 128), 256, 0, stream>>>(
      featsH, qkvWH, qkv_b, nullptr, geo_w, qH, kH, vTH, 3 * HID_, HID_, 2);

  // 5) fused attention
  (void)hipFuncSetAttribute((const void*)k_attn,
                            hipFuncAttributeMaxDynamicSharedMemorySize, SMEM_ATTN);
  k_attn<<<dim3(B_ * HEADS_, N_ / 32), 256, SMEM_ATTN, stream>>>(
      qH, kH, vTH, adj, geo_w, attOutH);

  // 6) ao GEMM + residual LN
  k_gemm<<<dim3(HID_ / 64, ROWS_ / 128), 256, 0, stream>>>(
      attOutH, aoWH, ao_b, aoTmp, nullptr, nullptr, nullptr, nullptr,
      HID_, HID_, 0);
  k_ln<<<ROWS_, HID_, 0, stream>>>(aoTmp, featsF, an_g, an_b, feats2F, feats2H, 0);

  // 7) heads: h0 (hp) and g1(relu)
  k_gemm<<<dim3(H2_ / 64, ROWS_ / 128), 256, 0, stream>>>(
      feats2H, hpWH, hp_b, h0F, nullptr, nullptr, nullptr, nullptr,
      H2_, HID_, 0);
  k_gemm<<<dim3(H4_ / 64, ROWS_ / 128), 256, 0, stream>>>(
      feats2H, g1WH, g1_b, g1F, nullptr, nullptr, nullptr, nullptr,
      H4_, HID_, 1);

  // 8) sequential GRU decoder (WMMA per step)
  k_gru<<<ROWS_ / 16, 128, 0, stream>>>(h0F, x, whhWH, op1WH,
                                        gru_wih, gru_bih, gru_bhh,
                                        op1_b, op2_w, op2_b, gruPred);

  // 9) final combine (gate/decay/hw) -> (B, HORIZON, N) + trailing scalar
  k_final<<<ROWS_ / 256, 256, 0, stream>>>(x, g1F, g2_w, g2_b, gruPred,
                                           log_decay, hw_w, hw_b, hw_ratio,
                                           (float*)d_out);
}
